// Steerable_2D_58858231824814
// MI455X (gfx1250) — matrix operand order repaired and checked by reading, not compiled, not run on hardware
//
#include <hip/hip_runtime.h>
#include <hip/hip_bf16.h>

typedef __attribute__((ext_vector_type(16))) _Float16 v16h;
typedef __attribute__((ext_vector_type(8)))  float    v8f;

#define NV 4000

// sigmoid via native v_exp_f32 + v_rcp_f32 (no IEEE division chain):
// 3 VALU ops instead of ~9.  ~1ulp rcp error << f16 WMMA quantization.
__device__ __forceinline__ float sigf(float x) {
  return __builtin_amdgcn_rcpf(1.0f + __expf(-x));
}
__device__ __forceinline__ int imin(int a, int b) { return a < b ? a : b; }
__device__ __forceinline__ int imax(int a, int b) { return a > b ? a : b; }

// ---------------------------------------------------------------------------
// Kernel 1: n-independent constants.
//   S1[c]  = sum_{c0} w1[c0,c]
//   cA[c]  = sigmoid(adj1*S1[c] + b1[c])   (off-diag F1 where A7=1)
//   c0[c]  = sigmoid(b1[c])                (off-diag F1 where A7=0)
//   GA/G0/S2[o] = contractions with w2
//   offsum[o] = sum_{i!=j in 13x13} sigmoid(cnt(i,j)*{GA|G0}[o]
//                                           + adj2*A13[i,j]*S2[o] + b2[o])
// Also zeroes the g_repr accumulator (required: graph replay re-runs us).
// ---------------------------------------------------------------------------
__global__ void __launch_bounds__(128)
prep_kernel(const float* __restrict__ w1, const float* __restrict__ b1,
            const float* __restrict__ adj1, const float* __restrict__ w2,
            const float* __restrict__ b2, const float* __restrict__ adj2,
            float* __restrict__ offsum, float* __restrict__ gacc) {
  __shared__ float sA[64];
  __shared__ float s0[64];
  const int t = threadIdx.x;  // o in [0,128)
  if (t < 64) {
    float s1 = 0.0f;
    for (int c0i = 0; c0i < 16; ++c0i) s1 += w1[c0i * 64 + t];
    sA[t] = sigf(adj1[0] * s1 + b1[t]);
    s0[t] = sigf(b1[t]);
  }
  __syncthreads();

  float GA = 0.0f, G0 = 0.0f, S2 = 0.0f;
  for (int c = 0; c < 64; ++c) {
    float w = w2[c * 128 + t];
    GA += sA[c] * w;
    G0 += s0[c] * w;
    S2 += w;
  }
  const float a2 = adj2[0];
  const float bb = b2[t];
  float off = 0.0f;
  for (int i = 0; i < 13; ++i) {
    for (int j = 0; j < 13; ++j) {
      if (i == j) continue;
      const int lo = imax(imax(i, j) - 9, -6);
      const int hi = imin(imin(i, j) - 3, 6);
      int cnt = hi - lo + 1;
      if (cnt < 0) cnt = 0;
      const int ad = (i > j) ? (i - j) : (j - i);
      float pre = bb;
      if (ad <= 3)      pre += (float)cnt * GA + a2 * S2;   // A13=1 band
      else if (ad <= 6) pre += (float)cnt * G0;
      // ad > 6: cnt==0, pre = b2[o]
      off += sigf(pre);
    }
  }
  offsum[t] = off;
  gacc[t] = 0.0f;
}

// ---------------------------------------------------------------------------
// Kernel 2: fused  M = sigmoid(labels@w1 + b1)  (WMMA, K=16 padded to 32)
//                  H = M@w2                     (WMMA, K=64 = 2 k-steps)
//                  g_diag partial = sum_rows  7*sig(7H+b2) + 2*sig(6H+b2)
//                                           + 2*sig(5H+b2) + 2*sig(4H+b2)
// One block = 16 rows of the 4000; 8 waves = 8 column tiles of H (128 cols).
// wave32 WMMA operand layouts per CDNA5 ISA 7.12.2.
// ---------------------------------------------------------------------------
__global__ void __launch_bounds__(256)
fused_kernel(const float* __restrict__ labels, const float* __restrict__ w1,
             const float* __restrict__ b1, const float* __restrict__ w2,
             const float* __restrict__ b2, float* __restrict__ gacc) {
  __shared__ _Float16 Mlds[16 * 64];
  __shared__ float part[128];

  const int row0 = blockIdx.x * 16;
  const int wave = threadIdx.x >> 5;   // 0..7
  const int lane = threadIdx.x & 31;
  const int lrow = lane & 15;
  const int lhi  = lane >> 4;          // 0 or 1
  const int ocol = wave * 16 + lrow;   // this lane's output column (fixed)

  if (threadIdx.x < 128) part[threadIdx.x] = 0.0f;
  __builtin_prefetch(w2 + (threadIdx.x << 5), 0, 0);  // global_prefetch_b8

  // ---- Phase 1: waves 0..3 build M tile (16x64, f16) in LDS ----
  if (wave < 4) {
    v16h a, b;
#pragma unroll
    for (int e = 0; e < 16; ++e) {
      const int Ka = e + (e >= 8 ? 8 : 0) + 8 * lhi;        // A: K in [0,32)
      a[e] = (Ka < 16) ? (_Float16)labels[(row0 + lrow) * 16 + Ka]
                       : (_Float16)0.0f;
      const int Kb = e + (e >= 8 ? 8 : 0) + 8 * lhi;        // B: K in [0,32)
      b[e] = (Kb < 16) ? (_Float16)w1[Kb * 64 + wave * 16 + lrow]
                       : (_Float16)0.0f;
    }
    v8f c = {};
    c = __builtin_amdgcn_wmma_f32_16x16x32_f16(false, a, false, b,
                                               (short)0, c, false, false);
    const float bias = b1[wave * 16 + lrow];
#pragma unroll
    for (int r = 0; r < 8; ++r) {
      const int mrow = r + 8 * lhi;
      Mlds[mrow * 64 + wave * 16 + lrow] = (_Float16)sigf(c[r] + bias);
    }
  }
  __syncthreads();

  // ---- Phase 2: H tile = M(16x64) @ w2[:, 16w:16w+16), K=64 ----
  v8f acc = {};
#pragma unroll
  for (int ks = 0; ks < 2; ++ks) {
    v16h a, b;
#pragma unroll
    for (int e = 0; e < 16; ++e) {
      const int Ka = ks * 32 + e + (e >= 8 ? 8 : 0) + 8 * lhi;
      a[e] = Mlds[lrow * 64 + Ka];
      const int Kb = ks * 32 + e + (e >= 8 ? 8 : 0) + 8 * lhi;
      b[e] = (_Float16)w2[Kb * 128 + ocol];
    }
    acc = __builtin_amdgcn_wmma_f32_16x16x32_f16(false, a, false, b,
                                                 (short)0, acc, false, false);
  }

  // ---- Epilogue: diagonal g_repr contribution, multiplicities 7/2/2/2 ----
  const float bb = b2[ocol];
  float s = 0.0f;
#pragma unroll
  for (int r = 0; r < 8; ++r) {
    const float h = acc[r];
    s += 7.0f * sigf(7.0f * h + bb)
       + 2.0f * sigf(6.0f * h + bb)
       + 2.0f * sigf(5.0f * h + bb)
       + 2.0f * sigf(4.0f * h + bb);
  }
  atomicAdd(&part[ocol], s);           // ds_add_f32, 2 lanes/col within wave
  __syncthreads();
  if (threadIdx.x < 128)
    atomicAdd(&gacc[threadIdx.x], part[threadIdx.x]);  // 1 global atomic/col
}

// ---------------------------------------------------------------------------
// Kernel 3: g_repr[o] = gacc[o] + N*offsum[o];  out = g_repr@fcw + fcb
// d_out layout (tuple order): [0] = output scalar, [1..128] = g_repr
// ---------------------------------------------------------------------------
__global__ void __launch_bounds__(128)
finalize_kernel(const float* __restrict__ gacc, const float* __restrict__ offsum,
                const float* __restrict__ fcw, const float* __restrict__ fcb,
                float* __restrict__ out) {
  __shared__ float red[128];
  const int o = threadIdx.x;
  const float g = gacc[o] + (float)NV * offsum[o];
  out[1 + o] = g;
  red[o] = g * fcw[o];
  __syncthreads();
  for (int st = 64; st > 0; st >>= 1) {
    if (o < st) red[o] += red[o + st];
    __syncthreads();
  }
  if (o == 0) out[0] = red[0] + fcb[0];
}

extern "C" void kernel_launch(void* const* d_in, const int* in_sizes, int n_in,
                              void* d_out, int out_size, void* d_ws, size_t ws_size,
                              hipStream_t stream) {
  (void)in_sizes; (void)n_in; (void)out_size; (void)ws_size;
  const float* labels = (const float*)d_in[0];
  const float* w1     = (const float*)d_in[1];
  const float* b1     = (const float*)d_in[2];
  const float* adj1   = (const float*)d_in[3];
  const float* w2     = (const float*)d_in[4];
  const float* b2     = (const float*)d_in[5];
  const float* adj2   = (const float*)d_in[6];
  const float* fcw    = (const float*)d_in[7];
  const float* fcb    = (const float*)d_in[8];
  float* out = (float*)d_out;

  float* offsum = (float*)d_ws;        // 128 floats
  float* gacc   = offsum + 128;        // 128 floats

  prep_kernel<<<1, 128, 0, stream>>>(w1, b1, adj1, w2, b2, adj2, offsum, gacc);
  fused_kernel<<<NV / 16, 256, 0, stream>>>(labels, w1, b1, w2, b2, gacc);
  finalize_kernel<<<1, 128, 0, stream>>>(gacc, offsum, fcw, fcb, out);
}
